// AttentionBlock_39333310496993
// MI455X (gfx1250) — matrix operand order repaired and checked by reading, not compiled, not run on hardware
//
#include <hip/hip_runtime.h>
#include <hip/hip_bf16.h>

#define B_ 4
#define C_ 128
#define T_ 4096
#define K_ 64
#define V_ 128
#define NT (T_/16)        // 256 i/j tiles
#define NC (T_/32)        // 128 j chunks (WMMA K-dim = 32 for bf16)

typedef __attribute__((ext_vector_type(16))) __bf16 v16bf;
typedef __attribute__((ext_vector_type(8)))  float  v8f;

union U16 {
    v16bf v;
    uint4 q[2];
};

__device__ __forceinline__ unsigned short f2bf(float f) {
    unsigned u = __builtin_bit_cast(unsigned, f);
    u += 0x7FFFu + ((u >> 16) & 1u);          // round-to-nearest-even
    return (unsigned short)(u >> 16);
}

// A-matrix (16x32, M x K) bf16 fragment, row-major source, row stride ld (elements).
// lane L: row = row0 + (L&15); k slots = {hi*8..hi*8+7, 16+hi*8..16+hi*8+7} + kk
__device__ __forceinline__ void loadA(U16& f, const unsigned short* base,
                                      int row0, int ld, int kk, int lane) {
    int r = lane & 15, hi = lane >> 4;
    const unsigned short* p = base + (size_t)(row0 + r) * ld + kk + hi * 8;
    f.q[0] = *(const uint4*)(p);
    f.q[1] = *(const uint4*)(p + 16);
}

// B-matrix (32x16, K x N) bf16 fragment, where B[k][n] = src[col0+n][kk+k]
// (src row-major with stride ld). lane L: col n=L&15; k slots = {hi*8..hi*8+7, 16+hi*8..16+hi*8+7}.
__device__ __forceinline__ void loadB(U16& f, const unsigned short* base,
                                      int col0, int ld, int kk, int lane) {
    int n = lane & 15, hi = lane >> 4;
    const unsigned short* p = base + (size_t)(col0 + n) * ld + kk + hi * 8;
    f.q[0] = *(const uint4*)(p);
    f.q[1] = *(const uint4*)(p + 16);
}

__device__ __forceinline__ v8f wmma_bf16(const U16& a, const U16& b, v8f c) {
    return __builtin_amdgcn_wmma_f32_16x16x32_bf16(false, a.v, false, b.v,
                                                   (short)0, c, false, false);
}

// ---------------------------------------------------------------------------
// Projection: Qs[b][t][64] = (x^T Wq + bq)/8 (bf16), Ks likewise (unscaled),
// Vt[b][v][t] = (x^T Wv + bv) transposed (bf16).
// Block: (b, 16-t tile); 256 threads, thread = output feature f (0..255).
// ---------------------------------------------------------------------------
__global__ __launch_bounds__(256) void proj_kernel(
    const float* __restrict__ x,
    const float* __restrict__ Wk, const float* __restrict__ bk,
    const float* __restrict__ Wq, const float* __restrict__ bq,
    const float* __restrict__ Wv, const float* __restrict__ bv,
    unsigned short* __restrict__ Qs, unsigned short* __restrict__ Ks,
    unsigned short* __restrict__ Vt) {
    int blk = blockIdx.x;
    int b = blk / (T_ / 16);
    int t0 = (blk % (T_ / 16)) * 16;
    int tid = threadIdx.x;

    __shared__ float xs[C_][16];
    for (int e = tid; e < C_ * 16; e += 256) {
        int c = e >> 4, tt = e & 15;
        xs[c][tt] = x[(size_t)(b * C_ + c) * T_ + t0 + tt];
    }
    __syncthreads();

    int f = tid;
    float bias = (f < 64) ? bq[f] : (f < 128) ? bk[f - 64] : bv[f - 128];
    float acc[16];
#pragma unroll
    for (int tt = 0; tt < 16; ++tt) acc[tt] = bias;

    for (int c = 0; c < C_; ++c) {
        float w = (f < 64)  ? Wq[c * 64 + f]
                : (f < 128) ? Wk[c * 64 + (f - 64)]
                            : Wv[c * 128 + (f - 128)];
#pragma unroll
        for (int tt = 0; tt < 16; ++tt) acc[tt] += xs[c][tt] * w;
    }

    if (f < 64) {
#pragma unroll
        for (int tt = 0; tt < 16; ++tt)
            Qs[(size_t)(b * T_ + t0 + tt) * K_ + f] = f2bf(acc[tt] * 0.125f);
    } else if (f < 128) {
#pragma unroll
        for (int tt = 0; tt < 16; ++tt)
            Ks[(size_t)(b * T_ + t0 + tt) * K_ + (f - 64)] = f2bf(acc[tt]);
    } else {
        unsigned short* dst = Vt + (size_t)(b * V_ + (f - 128)) * T_ + t0;
#pragma unroll
        for (int tt = 0; tt < 16; ++tt) dst[tt] = f2bf(acc[tt]);
    }
}

// Copy x -> out[:, :C, :] (identity part of concat)
__global__ __launch_bounds__(256) void copy_kernel(const float4* __restrict__ x4,
                                                   float4* __restrict__ out4) {
    int idx = blockIdx.x * 256 + threadIdx.x;
    const int per_b = C_ * T_ / 4;
    int b = idx / per_b, r = idx % per_b;
    out4[(size_t)b * ((C_ + V_) * T_ / 4) + r] = x4[idx];
}

// ---------------------------------------------------------------------------
// Pass 1: per-column softmax stats. Column j: m_j = max_{i<=j} S[i][j],
// Z_j = sum_{i<=j} exp(S[i][j]-m_j). One block per (b, j-tile). 8 waves
// split the i-tile range; BRANCH-FREE online logsumexp using a finite
// sentinel (-1e30): exp(sentinel - real) underflows to 0, and any bogus
// mass accumulated at the sentinel max is annihilated by the exp(m-m_new)
// weight when merged with a real max (every column has a real entry at i=j).
// ---------------------------------------------------------------------------
__global__ __launch_bounds__(256) void pass1_colstats(
    const unsigned short* __restrict__ Qs, const unsigned short* __restrict__ Ks,
    float* __restrict__ colM, float* __restrict__ colZ) {
    int blk = blockIdx.x;
    int b = blk >> 8, jt = blk & (NT - 1);
    int tid = threadIdx.x, wave = tid >> 5, lane = tid & 31;
    int n = lane & 15, hi = lane >> 4;

    const unsigned short* Qb = Qs + (size_t)b * T_ * K_;
    const unsigned short* Kb = Ks + (size_t)b * T_ * K_;

    U16 bk0, bk1;
    loadB(bk0, Kb, jt * 16, K_, 0, lane);
    loadB(bk1, Kb, jt * 16, K_, 32, lane);

    const float SNT = -1e30f;   // finite sentinel (== "-inf" for our ranges)
    int jg = jt * 16 + n;
    float m_run = SNT, s_run = 0.f;

    for (int it = wave; it <= jt; it += 8) {
        U16 a0, a1;
        loadA(a0, Qb, it * 16, K_, 0, lane);
        loadA(a1, Qb, it * 16, K_, 32, lane);
        v8f S = {};
        S = wmma_bf16(a0, bk0, S);
        S = wmma_bf16(a1, bk1, S);
        if (it == jt) {
#pragma unroll
            for (int e = 0; e < 8; ++e) {
                int mg = it * 16 + hi * 8 + e;
                if (mg > jg) S[e] = SNT;   // j < i masked
            }
        }
        float tm = S[0];
#pragma unroll
        for (int e = 1; e < 8; ++e) tm = fmaxf(tm, S[e]);
        float nm = fmaxf(m_run, tm);
        float acc = s_run * __expf(m_run - nm);
#pragma unroll
        for (int e = 0; e < 8; ++e) acc += __expf(S[e] - nm);
        s_run = acc; m_run = nm;
    }

    float m2 = __shfl_xor(m_run, 16, 32);
    float s2 = __shfl_xor(s_run, 16, 32);
    float mm = fmaxf(m_run, m2);
    float ss = s_run * __expf(m_run - mm) + s2 * __expf(m2 - mm);

    __shared__ float wm[8][16], wsv[8][16];
    if (lane < 16) { wm[wave][n] = mm; wsv[wave][n] = ss; }
    __syncthreads();
    if (tid < 16) {
        float M = SNT, Z = 0.f;
#pragma unroll
        for (int w = 0; w < 8; ++w) {
            float mw = wm[w][tid], sw = wsv[w][tid];
            float nm = fmaxf(M, mw);
            Z = Z * __expf(M - nm) + sw * __expf(mw - nm);
            M = nm;
        }
        colM[(size_t)b * T_ + jt * 16 + tid] = M;
        colZ[(size_t)b * T_ + jt * 16 + tid] = Z;
    }
}

// ---------------------------------------------------------------------------
// Pass 2 chunk: 32 j's = two 16-col S tiles -> P (bf16 via LDS) -> 8 PV WMMAs.
// V fragments are loaded first so their global loads overlap the S WMMAs,
// the exp VALU work, and the LDS round-trip. MASKED=true only for the single
// chunk containing the diagonal; the hot loop carries no compares.
// ---------------------------------------------------------------------------
template <bool MASKED>
__device__ __forceinline__ void process_chunk(
    int jc, int it, int lane,
    const unsigned short* __restrict__ Kb, const unsigned short* __restrict__ Vb,
    const float* __restrict__ cM, const float* __restrict__ cZ,
    const U16& aq0, const U16& aq1,
    unsigned short (*Pl)[32],      // this wave's 16x32 P staging tile
    v8f acc[8]) {
    int n = lane & 15, hi = lane >> 4;
    int j0 = jc * 32;

    U16 vb[8];
#pragma unroll
    for (int vt = 0; vt < 8; ++vt)
        loadB(vb[vt], Vb, vt * 16, T_, j0, lane);   // B[k][n] = Vt[vt*16+n][j0+k]

#pragma unroll
    for (int t2 = 0; t2 < 2; ++t2) {
        int jt = 2 * jc + t2;
        if (MASKED && jt < it) {
#pragma unroll
            for (int e = 0; e < 8; ++e) Pl[hi * 8 + e][t2 * 16 + n] = 0;
        } else {
            U16 b0, b1;
            loadB(b0, Kb, jt * 16, K_, 0, lane);
            loadB(b1, Kb, jt * 16, K_, 32, lane);
            v8f S = {};
            S = wmma_bf16(aq0, b0, S);
            S = wmma_bf16(aq1, b1, S);
            int jg = jt * 16 + n;
            float Mj = cM[jg];
            float iZ = 1.0f / cZ[jg];
#pragma unroll
            for (int e = 0; e < 8; ++e) {
                float p = __expf(S[e] - Mj) * iZ;
                if (MASKED) {
                    int mg = it * 16 + hi * 8 + e;
                    if (jg < mg) p = 0.f;           // j < i masked
                }
                Pl[hi * 8 + e][t2 * 16 + n] = f2bf(p);
            }
        }
    }
    // wave-local LDS write -> read hazard (DS ops are in-order per wave)
    asm volatile("s_wait_dscnt 0x0" ::: "memory");
    __builtin_amdgcn_wave_barrier();

    U16 pa;
    loadA(pa, &Pl[0][0], 0, 32, 0, lane);
#pragma unroll
    for (int vt = 0; vt < 8; ++vt)
        acc[vt] = wmma_bf16(pa, vb[vt], acc[vt]);
}

// One block per (b, i-tile); 4 waves split j-chunks, each accumulates a
// 16x128 fp32 tile (8 WMMA accumulators), reduced via LDS ds_add_f32.
__global__ __launch_bounds__(128) void pass2_attend(
    const unsigned short* __restrict__ Qs, const unsigned short* __restrict__ Ks,
    const unsigned short* __restrict__ Vt,
    const float* __restrict__ colM, const float* __restrict__ colZ,
    float* __restrict__ out) {
    int blk = blockIdx.x;
    int b = blk >> 8, it = blk & (NT - 1);
    int tid = threadIdx.x, wave = tid >> 5, lane = tid & 31;
    int n = lane & 15, hi = lane >> 4;

    __shared__ unsigned short Plds[4][16][32];
    __shared__ float Osm[16][V_];
    for (int e = tid; e < 16 * V_; e += 128) ((float*)Osm)[e] = 0.f;
    __syncthreads();

    const unsigned short* Qb = Qs + (size_t)b * T_ * K_;
    const unsigned short* Kb = Ks + (size_t)b * T_ * K_;
    const unsigned short* Vb = Vt + (size_t)b * V_ * T_;
    const float* cM = colM + (size_t)b * T_;
    const float* cZ = colZ + (size_t)b * T_;

    U16 aq0, aq1;
    loadA(aq0, Qb, it * 16, K_, 0, lane);
    loadA(aq1, Qb, it * 16, K_, 32, lane);

    v8f acc[8];
#pragma unroll
    for (int vt = 0; vt < 8; ++vt) acc[vt] = (v8f){};

    int jc0 = it >> 1;                       // chunk containing the diagonal
    if (wave == 0)
        process_chunk<true>(jc0, it, lane, Kb, Vb, cM, cZ, aq0, aq1,
                            Plds[0], acc);
    for (int jc = jc0 + 1 + wave; jc < NC; jc += 4)
        process_chunk<false>(jc, it, lane, Kb, Vb, cM, cZ, aq0, aq1,
                             Plds[wave], acc);

    // cross-wave reduction of the 16x128 tile
#pragma unroll
    for (int vt = 0; vt < 8; ++vt)
#pragma unroll
        for (int e = 0; e < 8; ++e)
            atomicAdd(&Osm[hi * 8 + e][vt * 16 + n], acc[vt][e]);
    __syncthreads();

    float* outb = out + (size_t)b * (C_ + V_) * T_;
    for (int e = tid; e < 16 * V_; e += 128) {
        int m = e & 15, v = e >> 4;
        outb[(size_t)(C_ + v) * T_ + it * 16 + m] = Osm[m][v];
    }
}

extern "C" void kernel_launch(void* const* d_in, const int* in_sizes, int n_in,
                              void* d_out, int out_size, void* d_ws, size_t ws_size,
                              hipStream_t stream) {
    const float* x  = (const float*)d_in[0];
    const float* Wk = (const float*)d_in[1];
    const float* bk = (const float*)d_in[2];
    const float* Wq = (const float*)d_in[3];
    const float* bq = (const float*)d_in[4];
    const float* Wv = (const float*)d_in[5];
    const float* bv = (const float*)d_in[6];
    float* out = (float*)d_out;

    unsigned short* Qs = (unsigned short*)d_ws;                 // B*T*64 bf16
    unsigned short* Ks = Qs + (size_t)B_ * T_ * K_;             // B*T*64 bf16
    unsigned short* Vt = Ks + (size_t)B_ * T_ * K_;             // B*V*T  bf16
    float* colM = (float*)(Vt + (size_t)B_ * V_ * T_);          // B*T f32
    float* colZ = colM + (size_t)B_ * T_;                       // B*T f32

    proj_kernel<<<B_ * (T_ / 16), 256, 0, stream>>>(x, Wk, bk, Wq, bq, Wv, bv,
                                                    Qs, Ks, Vt);
    copy_kernel<<<(B_ * C_ * T_ / 4) / 256, 256, 0, stream>>>(
        (const float4*)x, (float4*)out);
    pass1_colstats<<<B_ * NT, 256, 0, stream>>>(Qs, Ks, colM, colZ);
    pass2_attend<<<B_ * NT, 128, 0, stream>>>(Qs, Ks, Vt, colM, colZ, out);
}
